// MultiHeadAttention_48326972015056
// MI455X (gfx1250) — matrix operand order repaired and checked by reading, not compiled, not run on hardware
//
#include <hip/hip_runtime.h>
#include <cstdint>
#include <cstddef>

// ---- shapes (fixed by the reference) ----
static constexpr int Bn  = 2;
static constexpr int S   = 2048;
static constexpr int D   = 1024;
static constexpr int H   = 16;
static constexpr int HD  = 64;     // head dim
static constexpr int D3  = 3072;   // 3*D

typedef float  v8f   __attribute__((ext_vector_type(8)));
typedef float  v16f  __attribute__((ext_vector_type(16)));
typedef __bf16 v8bf  __attribute__((ext_vector_type(8)));
typedef __bf16 v16bf __attribute__((ext_vector_type(16)));

// -ln(10000)*2/64 for RoPE inverse frequencies
static constexpr float ROPE_C = -0.2878231366242558f;

// -----------------------------------------------------------------------------
// Kernel 1: qkv = x @ w_qkv^T + b_qkv, RoPE on q/k, scatter to
//   qb, kb : bf16 [B,H,S,HD]
//   vt     : bf16 [B,H,HD,S]   (transposed so P@V B-frags are contiguous)
// Tile: 256 rows x 64 cols. 8 waves, each wave = 32 rows (2 A-frags reuse each
// B-frag twice -> 8 WMMAs per 32-K chunk per wave).
// -----------------------------------------------------------------------------
__global__ __launch_bounds__(256)
void qkv_rope_kernel(const float* __restrict__ x,
                     const float* __restrict__ wqkv,
                     const float* __restrict__ bqkv,
                     __bf16* __restrict__ qb,
                     __bf16* __restrict__ kb,
                     __bf16* __restrict__ vt)
{
    __shared__ float tile[256 * 66];   // 67.6 KB

    const int tid  = threadIdx.x;
    const int lane = tid & 31;
    const int wave = tid >> 5;
    const int h16  = lane >> 4;
    const int ln   = lane & 15;
    const int rowBase = blockIdx.x * 256;   // over B*S = 4096
    const int nBase   = blockIdx.y * 64;    // over 3D  = 3072

    v8f acc[2][4] = {};

    const float* xrow0 = x + (size_t)(rowBase + wave * 32 + ln) * D;
    const float* xrow1 = xrow0 + (size_t)16 * D;

    for (int k0 = 0; k0 < D; k0 += 32) {
        // Two A fragments (16x32 bf16 each): per lane two 8-elem K chunks
        v16bf a[2];
        {
            v8f l0 = *(const v8f*)(xrow0 + k0 + h16 * 16);
            v8f u0 = *(const v8f*)(xrow0 + k0 + h16 * 16 + 8);
            v8f l1 = *(const v8f*)(xrow1 + k0 + h16 * 16);
            v8f u1 = *(const v8f*)(xrow1 + k0 + h16 * 16 + 8);
            v8bf lb0 = __builtin_convertvector(l0, v8bf);
            v8bf ub0 = __builtin_convertvector(u0, v8bf);
            v8bf lb1 = __builtin_convertvector(l1, v8bf);
            v8bf ub1 = __builtin_convertvector(u1, v8bf);
            #pragma unroll
            for (int i = 0; i < 8; ++i) {
                a[0][i] = lb0[i]; a[0][i + 8] = ub0[i];
                a[1][i] = lb1[i]; a[1][i + 8] = ub1[i];
            }
        }
        #pragma unroll
        for (int sub = 0; sub < 4; ++sub) {
            const int n = nBase + sub * 16 + ln;   // col of qkv = row of w_qkv
            v16f wv = *(const v16f*)(wqkv + (size_t)n * D + k0 + h16 * 16);
            v16bf b = __builtin_convertvector(wv, v16bf);
            acc[0][sub] = __builtin_amdgcn_wmma_f32_16x16x32_bf16(
                false, a[0], false, b, (short)0, acc[0][sub], false, false);
            acc[1][sub] = __builtin_amdgcn_wmma_f32_16x16x32_bf16(
                false, a[1], false, b, (short)0, acc[1][sub], false, false);
        }
    }

    // C layout: VGPR r, lanes0-15 -> M=r, lanes16-31 -> M=r+8, N=ln
    #pragma unroll
    for (int g = 0; g < 2; ++g)
        #pragma unroll
        for (int sub = 0; sub < 4; ++sub)
            #pragma unroll
            for (int r = 0; r < 8; ++r)
                tile[(wave * 32 + g * 16 + r + 8 * h16) * 66 + sub * 16 + ln]
                    = acc[g][sub][r];
    __syncthreads();

    // Epilogue: bias + RoPE per (even,odd) pair, scatter bf16. 8192 pairs/block.
    #pragma unroll
    for (int t = 0; t < 32; ++t) {
        const int pairIdx = t * 256 + tid;
        const int row = pairIdx >> 5;         // 0..255
        const int pr  = pairIdx & 31;         // pair within 64-col tile
        const float x0 = tile[row * 66 + 2 * pr];
        const float x1 = tile[row * 66 + 2 * pr + 1];
        const int grow = rowBase + row;
        const int b_ = grow >> 11;            // /S
        const int s_ = grow & 2047;           // %S
        const int nglob = nBase + 2 * pr;
        float v0 = x0 + bqkv[nglob];
        float v1 = x1 + bqkv[nglob + 1];
        const int sec = nglob >> 10;          // 0=q 1=k 2=v
        const int c   = nglob & 1023;
        const int h   = c >> 6;
        const int d0  = c & 63;
        if (sec < 2) {
            const int p = d0 >> 1;
            const float inv = __expf(ROPE_C * (float)p);
            const float ang = (float)s_ * inv;
            const float cs = __cosf(ang), sn = __sinf(ang);
            const float r0 = v0 * cs - v1 * sn;
            const float r1 = v0 * sn + v1 * cs;
            __bf16* dst = (sec == 0 ? qb : kb)
                          + (((size_t)b_ * H + h) * S + s_) * HD + d0;
            dst[0] = (__bf16)r0;
            dst[1] = (__bf16)r1;
        } else {
            __bf16* dst = vt + (((size_t)b_ * H + h) * HD + d0) * S + s_;
            dst[0] = (__bf16)v0;
            dst[S] = (__bf16)v1;   // next hd row
        }
    }
}

// async copy of 64 contiguous bytes global -> LDS (per calling lane)
__device__ __forceinline__
void async_copy64(unsigned lds_dst, const void* gsrc)
{
    asm volatile(
        "global_load_async_to_lds_b128 %0, %1, off\n\t"
        "global_load_async_to_lds_b128 %0, %1, off offset:16\n\t"
        "global_load_async_to_lds_b128 %0, %1, off offset:32\n\t"
        "global_load_async_to_lds_b128 %0, %1, off offset:48"
        :: "v"(lds_dst), "v"(gsrc) : "memory");
}

// -----------------------------------------------------------------------------
// Kernel 2: flash attention per (b,h). 64 queries/block, stream 64-key blocks
// with causal mask + online softmax. 4 waves, each owns 16 query rows.
// K/V blocks staged to LDS with GLOBAL_LOAD_ASYNC_TO_LDS_B128 (ASYNCcnt);
// next block prefetched into L2 while current tile runs on the WMMA units.
// -----------------------------------------------------------------------------
__global__ __launch_bounds__(128)
void flash_attn_kernel(const __bf16* __restrict__ qb,
                       const __bf16* __restrict__ kb,
                       const __bf16* __restrict__ vt,
                       __bf16* __restrict__ attn)
{
    __shared__ __bf16 Kbuf[64 * 80];       // [key][hd], padded stride 80
    __shared__ __bf16 Vbuf[64 * 80];       // [hd][key], padded stride 80
    __shared__ __bf16 Pbuf[4 * 16 * 80];   // per-wave 16x64 P tile

    const int tid  = threadIdx.x;
    const int lane = tid & 31;
    const int wave = tid >> 5;
    const int h16  = lane >> 4;
    const int ln   = lane & 15;
    const int bh     = blockIdx.y;        // b*H + h
    const int qBlock = blockIdx.x;
    const int qRow0  = qBlock * 64 + wave * 16;

    const __bf16* qbase = qb + (size_t)bh * S * HD;
    const __bf16* kbase = kb + (size_t)bh * S * HD;
    const __bf16* vbase = vt + (size_t)bh * HD * S;

    // Q fragments: A-layout, 2 K-chunks of 32 over hd=64
    v16bf qa[2];
    {
        const __bf16* qrow = qbase + (size_t)(qRow0 + ln) * HD;
        #pragma unroll
        for (int cki = 0; cki < 2; ++cki) {
            v8bf lo = *(const v8bf*)(qrow + cki * 32 + h16 * 16);
            v8bf hi = *(const v8bf*)(qrow + cki * 32 + h16 * 16 + 8);
            #pragma unroll
            for (int i = 0; i < 8; ++i) { qa[cki][i] = lo[i]; qa[cki][i + 8] = hi[i]; }
        }
    }

    v8f o[4] = {};
    float m[8], l[8];
    #pragma unroll
    for (int r = 0; r < 8; ++r) { m[r] = -1e30f; l[r] = 0.0f; }

    __bf16* Pw = Pbuf + wave * 16 * 80;
    const int srow  = tid >> 1;     // staging row 0..63
    const int shalf = tid & 1;      // staging 32-elem half

    for (int j = 0; j <= qBlock; ++j) {
        __syncthreads();
        // async staging: 128 threads x 64B each for K and V blocks
        async_copy64((unsigned)(uintptr_t)(Kbuf + srow * 80 + shalf * 32),
                     kbase + (size_t)(j * 64 + srow) * HD + shalf * 32);
        async_copy64((unsigned)(uintptr_t)(Vbuf + srow * 80 + shalf * 32),
                     vbase + (size_t)srow * S + j * 64 + shalf * 32);
        if (j < qBlock) {   // warm L2 for next block while we compute
            __builtin_prefetch(kbase + (size_t)((j + 1) * 64 + srow) * HD, 0, 1);
            __builtin_prefetch(vbase + (size_t)srow * S + (j + 1) * 64, 0, 1);
        }
        asm volatile("s_wait_asynccnt 0" ::: "memory");
        __syncthreads();

        // scores tile: 16 x 64 = 4 N-subtiles, 2 K-chunks each
        v8f sc[4] = {};
        #pragma unroll
        for (int cki = 0; cki < 2; ++cki) {
            #pragma unroll
            for (int sub = 0; sub < 4; ++sub) {
                const __bf16* kr = Kbuf + (sub * 16 + ln) * 80 + cki * 32 + h16 * 16;
                v16bf bfrag = *(const v16bf*)kr;
                sc[sub] = __builtin_amdgcn_wmma_f32_16x16x32_bf16(
                    false, qa[cki], false, bfrag, (short)0, sc[sub], false, false);
            }
        }

        // causal mask + online softmax (row stats reduced across 16-lane halves)
        const int kB = j * 64;
        #pragma unroll
        for (int r = 0; r < 8; ++r) {
            const int qrow = qRow0 + r + 8 * h16;
            float vmax = -1e30f;
            float val[4];
            #pragma unroll
            for (int sub = 0; sub < 4; ++sub) {
                const int tkey = kB + sub * 16 + ln;
                float v = sc[sub][r] * 0.125f;          // 1/sqrt(64)
                if (tkey > qrow) v = -1e30f;
                val[sub] = v;
                vmax = fmaxf(vmax, v);
            }
            #pragma unroll
            for (int mk = 1; mk < 16; mk <<= 1)
                vmax = fmaxf(vmax, __shfl_xor(vmax, mk, 32));
            const float mnew  = fmaxf(m[r], vmax);
            const float alpha = __expf(m[r] - mnew);
            float rsum = 0.0f;
            #pragma unroll
            for (int sub = 0; sub < 4; ++sub) {
                const float p = __expf(val[sub] - mnew);
                rsum += p;
                Pw[(r + 8 * h16) * 80 + sub * 16 + ln] = (__bf16)p;
            }
            #pragma unroll
            for (int mk = 1; mk < 16; mk <<= 1)
                rsum += __shfl_xor(rsum, mk, 32);
            l[r] = l[r] * alpha + rsum;
            m[r] = mnew;
            #pragma unroll
            for (int sub = 0; sub < 4; ++sub) o[sub][r] *= alpha;
        }
        // per-wave LDS transpose of P: DS is in-order per wave; wait data return
        asm volatile("s_wait_dscnt 0" ::: "memory");

        // O += P @ V   (K-dim = 64 keys, 2 chunks; N = hd, 4 subtiles)
        #pragma unroll
        for (int cki = 0; cki < 2; ++cki) {
            v16bf pa;
            const __bf16* prow = Pw + ln * 80 + cki * 32;
            v8bf plo = *(const v8bf*)(prow + h16 * 16);
            v8bf phi = *(const v8bf*)(prow + h16 * 16 + 8);
            #pragma unroll
            for (int i = 0; i < 8; ++i) { pa[i] = plo[i]; pa[i + 8] = phi[i]; }
            #pragma unroll
            for (int sub = 0; sub < 4; ++sub) {
                const __bf16* vr = Vbuf + (sub * 16 + ln) * 80 + cki * 32 + h16 * 16;
                v16bf vb = *(const v16bf*)vr;
                o[sub] = __builtin_amdgcn_wmma_f32_16x16x32_bf16(
                    false, pa, false, vb, (short)0, o[sub], false, false);
            }
        }
    }

    // normalize and write attn output bf16 as [B,S,D]
    const int b_ = bh >> 4, h_ = bh & 15;
    #pragma unroll
    for (int r = 0; r < 8; ++r) {
        const float inv = 1.0f / l[r];
        const int qrow = qRow0 + r + 8 * h16;
        __bf16* orow = attn + ((size_t)b_ * S + qrow) * D + h_ * HD;
        #pragma unroll
        for (int sub = 0; sub < 4; ++sub)
            orow[sub * 16 + ln] = (__bf16)(o[sub][r] * inv);
    }
}

// -----------------------------------------------------------------------------
// Kernel 3: out = attn @ w_out^T + b_out  (fp32 result)
// 256x64 tiling; each wave = 32 rows so every B fragment feeds 2 WMMAs.
// -----------------------------------------------------------------------------
__global__ __launch_bounds__(256)
void out_proj_kernel(const __bf16* __restrict__ attn,
                     const float* __restrict__ wout,
                     const float* __restrict__ bout,
                     float* __restrict__ out)
{
    const int tid  = threadIdx.x;
    const int lane = tid & 31;
    const int wave = tid >> 5;
    const int h16  = lane >> 4;
    const int ln   = lane & 15;
    const int rowBase = blockIdx.x * 256;
    const int nBase   = blockIdx.y * 64;

    v8f acc[2][4] = {};
    const __bf16* arow0 = attn + (size_t)(rowBase + wave * 32 + ln) * D;
    const __bf16* arow1 = arow0 + (size_t)16 * D;

    for (int k0 = 0; k0 < D; k0 += 32) {
        v16bf a[2];
        {
            v8bf l0 = *(const v8bf*)(arow0 + k0 + h16 * 16);
            v8bf u0 = *(const v8bf*)(arow0 + k0 + h16 * 16 + 8);
            v8bf l1 = *(const v8bf*)(arow1 + k0 + h16 * 16);
            v8bf u1 = *(const v8bf*)(arow1 + k0 + h16 * 16 + 8);
            #pragma unroll
            for (int i = 0; i < 8; ++i) {
                a[0][i] = l0[i]; a[0][i + 8] = u0[i];
                a[1][i] = l1[i]; a[1][i + 8] = u1[i];
            }
        }
        #pragma unroll
        for (int sub = 0; sub < 4; ++sub) {
            const int n = nBase + sub * 16 + ln;
            v16f wv = *(const v16f*)(wout + (size_t)n * D + k0 + h16 * 16);
            v16bf b = __builtin_convertvector(wv, v16bf);
            acc[0][sub] = __builtin_amdgcn_wmma_f32_16x16x32_bf16(
                false, a[0], false, b, (short)0, acc[0][sub], false, false);
            acc[1][sub] = __builtin_amdgcn_wmma_f32_16x16x32_bf16(
                false, a[1], false, b, (short)0, acc[1][sub], false, false);
        }
    }

    #pragma unroll
    for (int sub = 0; sub < 4; ++sub) {
        const int n = nBase + sub * 16 + ln;
        const float bb = bout[n];
        #pragma unroll
        for (int g = 0; g < 2; ++g)
            #pragma unroll
            for (int r = 0; r < 8; ++r) {
                const int row = rowBase + wave * 32 + g * 16 + r + 8 * h16;
                out[(size_t)row * D + n] = acc[g][sub][r] + bb;
            }
    }
}

// -----------------------------------------------------------------------------
extern "C" void kernel_launch(void* const* d_in, const int* in_sizes, int n_in,
                              void* d_out, int out_size, void* d_ws, size_t ws_size,
                              hipStream_t stream)
{
    const float* x    = (const float*)d_in[0];
    const float* wqkv = (const float*)d_in[1];
    const float* bqkv = (const float*)d_in[2];
    const float* wout = (const float*)d_in[3];
    const float* bout = (const float*)d_in[4];
    float* out = (float*)d_out;

    const size_t perBuf = (size_t)Bn * H * S * HD;   // 4,194,304 elems (8 MB bf16)
    __bf16* qb   = (__bf16*)d_ws;
    __bf16* kb   = qb + perBuf;
    __bf16* vt   = kb + perBuf;
    __bf16* attn = vt + perBuf;

    // K1: (B*S/256) x (3D/64) = 16 x 48 blocks, 256 thr (8 waves)
    qkv_rope_kernel<<<dim3(16, 48), 256, 0, stream>>>(x, wqkv, bqkv, qb, kb, vt);
    // K2: (S/64) x (B*H) = 32 x 32 blocks, 128 thr (4 waves)
    flash_attn_kernel<<<dim3(32, 32), 128, 0, stream>>>(qb, kb, vt, attn);
    // K3: (B*S/256) x (D/64) = 16 x 16 blocks, 256 thr
    out_proj_kernel<<<dim3(16, 16), 256, 0, stream>>>(attn, wout, bout, out);
}
